// MultiHeadAttention_78237124264537
// MI455X (gfx1250) — matrix operand order repaired and checked by reading, not compiled, not run on hardware
//
#include <hip/hip_runtime.h>
#include <hip/hip_bf16.h>

typedef __attribute__((ext_vector_type(16))) __bf16 v16bf;
typedef __attribute__((ext_vector_type(8)))  __bf16 v8bf;
typedef __attribute__((ext_vector_type(8)))  float  v8f;

typedef unsigned int u32x4 __attribute__((ext_vector_type(4)));
typedef int          i32x8 __attribute__((ext_vector_type(8)));
typedef int          i32x4 __attribute__((ext_vector_type(4)));

#define D_MODEL 1024
#define N_HEADS 16
#define DK      64
#define SEQ     2048
#define BATCH   2
#define NEG_INF (-1e9f)

#if __has_builtin(__builtin_amdgcn_tensor_load_to_lds)
#define HAVE_TDM 1
#else
#define HAVE_TDM 0
#endif

static __device__ inline v8f zero_v8f() {
  v8f z;
#pragma unroll
  for (int i = 0; i < 8; ++i) z[i] = 0.0f;
  return z;
}

static __device__ inline v8f wmma_bf16(v16bf a, v16bf b, v8f c) {
  return __builtin_amdgcn_wmma_f32_16x16x32_bf16(false, a, false, b, (short)0, c,
                                                 false, false);
}

// B-fragment: 16 consecutive bf16 (K-dim) for one column (this code's contiguous k = 16h + s, same as the A side)
static __device__ inline v16bf load16_bf(const __bf16* p) {
  v8bf lo = *(const v8bf*)(p);
  v8bf hi = *(const v8bf*)(p + 8);
  v16bf r;
#pragma unroll
  for (int i = 0; i < 8; ++i) { r[i] = lo[i]; r[i + 8] = hi[i]; }
  return r;
}

// A-fragment: two 8-element chunks at K-offsets c1 / c2 within a row
static __device__ inline v16bf loadA_bf(const __bf16* rowp, int c1, int c2) {
  v8bf lo = *(const v8bf*)(rowp + c1);
  v8bf hi = *(const v8bf*)(rowp + c2);
  v16bf r;
#pragma unroll
  for (int i = 0; i < 8; ++i) { r[i] = lo[i]; r[i + 8] = hi[i]; }
  return r;
}

static __device__ inline void wait_tensorcnt0() {
#if __has_builtin(__builtin_amdgcn_s_wait_tensorcnt)
  __builtin_amdgcn_s_wait_tensorcnt(0);
#else
  asm volatile("s_wait_tensorcnt 0x0" ::: "memory");
#endif
}

#if HAVE_TDM
// 2-D TDM descriptor load: tile (tile_x x tile_y) of bf16 from a row-major
// tensor (row length stride_x elements) into contiguous LDS at lds_off.
static __device__ inline void tdm_load_2d(unsigned lds_off, const __bf16* gptr,
                                          unsigned tile_x, unsigned tile_y,
                                          unsigned tensor_x, unsigned tensor_y,
                                          unsigned stride_x) {
  unsigned long long ga = (unsigned long long)(uintptr_t)gptr;
  u32x4 g0;
  g0[0] = 1u;                                      // count=1 (valid), user mode
  g0[1] = lds_off;                                 // LDS byte address
  g0[2] = (unsigned)(ga & 0xffffffffu);            // global_addr[31:0]
  g0[3] = (unsigned)((ga >> 32) & 0x01ffffffu) | (2u << 30);  // [56:32] | type=2
  i32x8 g1;
  g1[0] = (int)(1u << 16);                         // data_size=1 (2 bytes)
  g1[1] = (int)((tensor_x & 0xffffu) << 16);       // tensor_dim0[15:0]
  g1[2] = (int)(((tensor_x >> 16) & 0xffffu) | ((tensor_y & 0xffffu) << 16));
  g1[3] = (int)(((tensor_y >> 16) & 0xffffu) | ((tile_x & 0xffffu) << 16));
  g1[4] = (int)(tile_y & 0xffffu);                 // tile_dim1 ; tile_dim2=0
  g1[5] = (int)stride_x;                           // tensor_dim0_stride[31:0]
  g1[6] = 0;
  g1[7] = 0;
  i32x4 z4 = {0, 0, 0, 0};
  i32x8 z8 = {0, 0, 0, 0, 0, 0, 0, 0};
  // 6-arg toolchain form: groups 0..3 + trailing group slot + cache policy
  __builtin_amdgcn_tensor_load_to_lds(g0, g1, z4, z4, z8, 0);
}
#endif

// ---------------------------------------------------------------------------
// Kernel 0: elementwise f32 -> bf16 (8 elems/thread, b128 in / b128 out)
// ---------------------------------------------------------------------------
__global__ void cvt_f32_bf16_kernel(const float* __restrict__ src,
                                    __bf16* __restrict__ dst, int n8) {
  const int i = blockIdx.x * blockDim.x + threadIdx.x;
  if (i >= n8) return;
  const float4* s4 = (const float4*)src;
  const float4 a = s4[(size_t)i * 2];
  const float4 b = s4[(size_t)i * 2 + 1];
  v8bf o;
  o[0] = (__bf16)a.x; o[1] = (__bf16)a.y; o[2] = (__bf16)a.z; o[3] = (__bf16)a.w;
  o[4] = (__bf16)b.x; o[5] = (__bf16)b.y; o[6] = (__bf16)b.z; o[7] = (__bf16)b.w;
  *(v8bf*)(dst + (size_t)i * 8) = o;
}

// ---------------------------------------------------------------------------
// Kernel 1: fused QKV projection (bf16 x bf16 -> f32 acc -> bf16).
//   q,k -> [B,H,S,64] ; v -> [B,H,64,S] (transposed). 32x64 tile per wave.
// ---------------------------------------------------------------------------
__global__ void qkv_proj_kernel(const __bf16* __restrict__ Xq,
                                const __bf16* __restrict__ Xk,
                                const __bf16* __restrict__ Xv,
                                const __bf16* __restrict__ WQ,
                                const __bf16* __restrict__ WK,
                                const __bf16* __restrict__ WV,
                                __bf16* __restrict__ qb,
                                __bf16* __restrict__ kb,
                                __bf16* __restrict__ vtb) {
  const int mode = blockIdx.z;  // 0=Q 1=K 2=V
  const __bf16* X = (mode == 0) ? Xq : (mode == 1) ? Xk : Xv;
  const __bf16* W = (mode == 0) ? WQ : (mode == 1) ? WK : WV;

  const int lane = threadIdx.x & 31;
  const int ln = lane & 15, h2 = lane >> 4;
  const int c1 = 16 * h2, c2 = 16 * h2 + 8;
  const int m0 = blockIdx.x * 32;
  const int c0 = blockIdx.y * 64;

  v8f acc[2][4];
#pragma unroll
  for (int i = 0; i < 2; ++i)
#pragma unroll
    for (int j = 0; j < 4; ++j) acc[i][j] = zero_v8f();

  for (int kk = 0; kk < D_MODEL; kk += 32) {
    v16bf a0 = loadA_bf(X + (size_t)(m0 + ln) * D_MODEL + kk, c1, c2);
    v16bf a1 = loadA_bf(X + (size_t)(m0 + 16 + ln) * D_MODEL + kk, c1, c2);
#pragma unroll
    for (int j = 0; j < 4; ++j) {
      v16bf bf = load16_bf(W + (size_t)(c0 + 16 * j + ln) * D_MODEL + kk + 16 * h2);
      acc[0][j] = wmma_bf16(a0, bf, acc[0][j]);
      acc[1][j] = wmma_bf16(a1, bf, acc[1][j]);
    }
  }

#pragma unroll
  for (int i = 0; i < 2; ++i)
#pragma unroll
    for (int j = 0; j < 4; ++j)
#pragma unroll
      for (int r = 0; r < 8; ++r) {
        const int row = m0 + 16 * i + 8 * h2 + r;
        const int col = c0 + 16 * j + ln;
        const int b = row >> 11, s = row & 2047;
        const int h = col >> 6,  d = col & 63;
        const __bf16 v = (__bf16)acc[i][j][r];
        if (mode < 2) {
          __bf16* dst = (mode == 0) ? qb : kb;
          dst[(((size_t)b * N_HEADS + h) * SEQ + s) * DK + d] = v;
        } else {
          vtb[(((size_t)b * N_HEADS + h) * DK + d) * SEQ + s] = v;
        }
      }
}

// ---------------------------------------------------------------------------
// Kernel 2: flash attention. 8 waves/block, each owns a 16-query tile;
// K/V 32-key tiles shared in LDS, double-buffered via the Tensor Data Mover
// (wave 0 issues next tile while everyone computes the current one).
// ---------------------------------------------------------------------------
__global__ void __launch_bounds__(256)
flash_attn_kernel(const __bf16* __restrict__ qb,
                  const __bf16* __restrict__ kb,
                  const __bf16* __restrict__ vtb,
                  const unsigned char* __restrict__ mask,
                  __bf16* __restrict__ ctx) {
  __shared__ __align__(16) __bf16 ldsK[2][32][64];  // [buf][key][d]
  __shared__ __align__(16) __bf16 ldsV[2][64][32];  // [buf][d][key]
  __shared__ __align__(16) __bf16 ldsP[8][16][32];  // per-wave P relayout

  const int tid = threadIdx.x;
  const int wid = tid >> 5;
  const int lane = tid & 31;
  const int ln = lane & 15, h2 = lane >> 4;
  const int c1 = 16 * h2, c2 = 16 * h2 + 8;
  const int bh = blockIdx.y;
  const int b = bh >> 4, h = bh & 15;
  const int m0 = blockIdx.x * 128 + wid * 16;  // this wave's query tile

  const size_t baseq = (size_t)bh * SEQ;
  const __bf16* kbase = kb + baseq * DK;
  const __bf16* vbase = vtb + (size_t)bh * DK * SEQ;

  const __bf16* qrow = qb + (baseq + m0 + ln) * DK;
  const v16bf qa0 = loadA_bf(qrow, c1, c2);
  const v16bf qa1 = loadA_bf(qrow + 32, c1, c2);

  v8f o[4];
#pragma unroll
  for (int t = 0; t < 4; ++t) o[t] = zero_v8f();
  float m_run[8], l_run[8];
#pragma unroll
  for (int r = 0; r < 8; ++r) { m_run[r] = -3.0e38f; l_run[r] = 0.0f; }

  const unsigned char* mbase = mask + (size_t)b * SEQ * SEQ;
  const float scale = 0.125f;

#if HAVE_TDM
  const unsigned kofs = (unsigned)(uintptr_t)&ldsK[0][0][0];
  const unsigned vofs = (unsigned)(uintptr_t)&ldsV[0][0][0];
  if (wid == 0) {  // prefetch first tile into buffer 0
    tdm_load_2d(kofs, kbase, DK, 32, DK, SEQ, DK);
    tdm_load_2d(vofs, vbase, 32, DK, SEQ, DK, SEQ);
  }
#endif

  int buf = 0;
  for (int k0 = 0; k0 < SEQ; k0 += 32, buf ^= 1) {
#if HAVE_TDM
    if (wid == 0) wait_tensorcnt0();   // current tile landed in LDS
    __syncthreads();                   // everyone done with the other buffer
    if (wid == 0 && k0 + 32 < SEQ) {   // prefetch next tile into other buffer
      tdm_load_2d(kofs + (unsigned)(buf ^ 1) * 4096u,
                  kbase + (size_t)(k0 + 32) * DK, DK, 32, DK, SEQ, DK);
      tdm_load_2d(vofs + (unsigned)(buf ^ 1) * 4096u,
                  vbase + (k0 + 32), 32, DK, SEQ, DK, SEQ);
    }
#else
    __syncthreads();  // protect buffer reuse
    {                 // cooperative copy: 256 threads x 16B each per tile
      const int row8 = tid >> 3, seg8 = tid & 7;    // K: 32 rows x 8 segs
      *(v8bf*)&ldsK[buf][row8][seg8 * 8] =
          *(const v8bf*)(kbase + (size_t)(k0 + row8) * DK + seg8 * 8);
      const int row4 = tid >> 2, seg4 = tid & 3;    // V: 64 rows x 4 segs
      *(v8bf*)&ldsV[buf][row4][seg4 * 8] =
          *(const v8bf*)(vbase + (size_t)row4 * SEQ + k0 + seg4 * 8);
    }
    __syncthreads();
#endif

    // ---- scores: S = Q @ K^T ----
    v8f s0 = zero_v8f(), s1 = zero_v8f();
    {
      v16bf kf0 = load16_bf(&ldsK[buf][ln][16 * h2]);
      s0 = wmma_bf16(qa0, kf0, s0);
      v16bf kf1 = load16_bf(&ldsK[buf][ln][32 + 16 * h2]);
      s0 = wmma_bf16(qa1, kf1, s0);
      v16bf kf2 = load16_bf(&ldsK[buf][16 + ln][16 * h2]);
      s1 = wmma_bf16(qa0, kf2, s1);
      v16bf kf3 = load16_bf(&ldsK[buf][16 + ln][32 + 16 * h2]);
      s1 = wmma_bf16(qa1, kf3, s1);
    }

    // ---- online softmax (row r lives on 16 lanes of this half-wave) ----
#pragma unroll
    for (int r = 0; r < 8; ++r) {
      const int row = m0 + 8 * h2 + r;
      float v0 = s0[r] * scale;
      float v1 = s1[r] * scale;
      const unsigned char* mp = mbase + (size_t)row * SEQ + k0;
      if (mp[ln])      v0 = NEG_INF;
      if (mp[16 + ln]) v1 = NEG_INF;

      float cm = fmaxf(v0, v1);
      cm = fmaxf(cm, __shfl_xor(cm, 1));
      cm = fmaxf(cm, __shfl_xor(cm, 2));
      cm = fmaxf(cm, __shfl_xor(cm, 4));
      cm = fmaxf(cm, __shfl_xor(cm, 8));

      const float mn = fmaxf(m_run[r], cm);
      const float alpha = __expf(m_run[r] - mn);
      const float p0 = __expf(v0 - mn);
      const float p1 = __expf(v1 - mn);
      float rs = p0 + p1;
      rs += __shfl_xor(rs, 1);
      rs += __shfl_xor(rs, 2);
      rs += __shfl_xor(rs, 4);
      rs += __shfl_xor(rs, 8);

      l_run[r] = l_run[r] * alpha + rs;
      m_run[r] = mn;
#pragma unroll
      for (int t = 0; t < 4; ++t) o[t][r] *= alpha;

      ldsP[wid][8 * h2 + r][ln]      = (__bf16)p0;   // per-wave slice:
      ldsP[wid][8 * h2 + r][16 + ln] = (__bf16)p1;   // wave-ordered DS ops
    }

    // ---- O += P @ V ----
    v16bf pa = loadA_bf(&ldsP[wid][ln][0], c1, c2);
#pragma unroll
    for (int t = 0; t < 4; ++t) {
      v16bf vf = load16_bf(&ldsV[buf][t * 16 + ln][16 * h2]);
      o[t] = wmma_bf16(pa, vf, o[t]);
    }
  }

  // ---- normalize, store context bf16 [B,S,H*64] ----
#pragma unroll
  for (int r = 0; r < 8; ++r) {
    const int row = m0 + 8 * h2 + r;
    const float inv = 1.0f / l_run[r];
#pragma unroll
    for (int t = 0; t < 4; ++t) {
      ctx[((size_t)b * SEQ + row) * D_MODEL + h * DK + t * 16 + ln] =
          (__bf16)(o[t][r] * inv);
    }
  }
}

// ---------------------------------------------------------------------------
// Kernel 3: output projection  out = ctx @ W_O^T  (bf16 x bf16 -> f32)
// ---------------------------------------------------------------------------
__global__ void out_proj_kernel(const __bf16* __restrict__ ctx,
                                const __bf16* __restrict__ WO,
                                float* __restrict__ out) {
  const int lane = threadIdx.x & 31;
  const int ln = lane & 15, h2 = lane >> 4;
  const int c1 = 16 * h2, c2 = 16 * h2 + 8;
  const int m0 = blockIdx.x * 32;
  const int c0 = blockIdx.y * 64;

  v8f acc[2][4];
#pragma unroll
  for (int i = 0; i < 2; ++i)
#pragma unroll
    for (int j = 0; j < 4; ++j) acc[i][j] = zero_v8f();

  for (int kk = 0; kk < D_MODEL; kk += 32) {
    v16bf a0 = loadA_bf(ctx + (size_t)(m0 + ln) * D_MODEL + kk, c1, c2);
    v16bf a1 = loadA_bf(ctx + (size_t)(m0 + 16 + ln) * D_MODEL + kk, c1, c2);
#pragma unroll
    for (int j = 0; j < 4; ++j) {
      v16bf bf = load16_bf(WO + (size_t)(c0 + 16 * j + ln) * D_MODEL + kk + 16 * h2);
      acc[0][j] = wmma_bf16(a0, bf, acc[0][j]);
      acc[1][j] = wmma_bf16(a1, bf, acc[1][j]);
    }
  }

#pragma unroll
  for (int i = 0; i < 2; ++i)
#pragma unroll
    for (int j = 0; j < 4; ++j)
#pragma unroll
      for (int r = 0; r < 8; ++r) {
        const int row = m0 + 16 * i + 8 * h2 + r;
        const int col = c0 + 16 * j + ln;
        out[(size_t)row * D_MODEL + col] = acc[i][j][r];
      }
}

// ---------------------------------------------------------------------------
extern "C" void kernel_launch(void* const* d_in, const int* in_sizes, int n_in,
                              void* d_out, int out_size, void* d_ws,
                              size_t ws_size, hipStream_t stream) {
  const float* Qf  = (const float*)d_in[0];
  const float* Kf  = (const float*)d_in[1];
  const float* Vf  = (const float*)d_in[2];
  const unsigned char* mask = (const unsigned char*)d_in[3];
  const float* WQf = (const float*)d_in[4];
  const float* WKf = (const float*)d_in[5];
  const float* WVf = (const float*)d_in[6];
  const float* WOf = (const float*)d_in[7];
  float* out = (float*)d_out;

  const size_t PER = (size_t)BATCH * N_HEADS * SEQ * DK;   // 4,194,304
  const size_t WSZ = (size_t)D_MODEL * D_MODEL;            // 1,048,576
  __bf16* p = (__bf16*)d_ws;
  __bf16* qb  = p; p += PER;
  __bf16* kb  = p; p += PER;
  __bf16* vtb = p; p += PER;
  __bf16* ctx = p; p += PER;
  __bf16* xq  = p; p += PER;
  __bf16* xk  = p; p += PER;
  __bf16* xv  = p; p += PER;
  __bf16* wq  = p; p += WSZ;
  __bf16* wk  = p; p += WSZ;
  __bf16* wv  = p; p += WSZ;
  __bf16* wo  = p; p += WSZ;

  // 0) one-shot f32 -> bf16 conversions (memory-bound, 23.3 TB/s path)
  const int nx8 = (int)(PER / 8), nw8 = (int)(WSZ / 8);
  cvt_f32_bf16_kernel<<<(nx8 + 255) / 256, 256, 0, stream>>>(Qf,  xq, nx8);
  cvt_f32_bf16_kernel<<<(nx8 + 255) / 256, 256, 0, stream>>>(Kf,  xk, nx8);
  cvt_f32_bf16_kernel<<<(nx8 + 255) / 256, 256, 0, stream>>>(Vf,  xv, nx8);
  cvt_f32_bf16_kernel<<<(nw8 + 255) / 256, 256, 0, stream>>>(WQf, wq, nw8);
  cvt_f32_bf16_kernel<<<(nw8 + 255) / 256, 256, 0, stream>>>(WKf, wk, nw8);
  cvt_f32_bf16_kernel<<<(nw8 + 255) / 256, 256, 0, stream>>>(WVf, wv, nw8);
  cvt_f32_bf16_kernel<<<(nw8 + 255) / 256, 256, 0, stream>>>(WOf, wo, nw8);

  // 1) QKV projections
  qkv_proj_kernel<<<dim3((BATCH * SEQ) / 32, D_MODEL / 64, 3), 32, 0, stream>>>(
      xq, xk, xv, wq, wk, wv, qb, kb, vtb);

  // 2) flash attention: 16 query-blocks (8 waves each) x 32 (b,h)
  flash_attn_kernel<<<dim3(SEQ / 128, BATCH * N_HEADS), 256, 0, stream>>>(
      qb, kb, vtb, mask, ctx);

  // 3) output projection -> f32 d_out
  out_proj_kernel<<<dim3((BATCH * SEQ) / 32, D_MODEL / 64), 32, 0, stream>>>(
      ctx, wo, out);
}
